// Mamba_74294344286894
// MI455X (gfx1250) — compile-verified, matrix operand order checked by reading, not run on hardware
//
#include <hip/hip_runtime.h>
#include <hip/hip_bf16.h>
#include <math.h>

// ---- problem dimensions (fixed by the reference) ----
#define DIM     1024
#define DSTATE  16
#define DCONV   4
#define DI      2048           // DIM * EXPAND
#define DTRANK  64             // ceil(DIM/16)
#define BATCH   2
#define LSEQ    1024
#define NROWS   (BATCH * LSEQ)           // 2048 flattened (b,l) rows
#define XPROJ_E (DTRANK + 2 * DSTATE)    // 96

// ---- GEMM tiling ----
#define BM 128                 // block tile M
#define BN 64                  // block tile N
#define BK 32                  // K step (one WMMA depth)
#define LDSTR 40               // LDS row stride in bf16 elems (32 data + 8 pad = 80B)

typedef __attribute__((ext_vector_type(16))) __bf16 v16bf;
typedef __attribute__((ext_vector_type(8)))  __bf16 v8bf;
typedef __attribute__((ext_vector_type(8)))  float  v8f;

__device__ __forceinline__ float sigf(float x) { return 1.0f / (1.0f + __expf(-x)); }

// load 8 consecutive f32 from global, convert to 8 bf16
__device__ __forceinline__ v8bf cvt8(const float* __restrict__ g)
{
    float4 f0 = *(const float4*)g;
    float4 f1 = *(const float4*)(g + 4);
    v8bf r;
    r[0] = (__bf16)f0.x; r[1] = (__bf16)f0.y; r[2] = (__bf16)f0.z; r[3] = (__bf16)f0.w;
    r[4] = (__bf16)f1.x; r[5] = (__bf16)f1.y; r[6] = (__bf16)f1.z; r[7] = (__bf16)f1.w;
    return r;
}

// -----------------------------------------------------------------------------
// Blocked GEMM: C[M,N] = A[M,K] * B[N,K]^T, f32 in memory, bf16 WMMA math with
// f32 accumulation. Row-major with explicit leading dimensions.
//
// Block = 256 threads (8 waves, wave32). Block tile 128x64; waves in a 4x2
// grid each own a 32x32 region = 2x2 WMMA tiles (A-frags reused across N,
// B-frags across M). Per K-step of 32: cooperative f32->bf16 staging into
// padded LDS, then fragments read back as 16B ds loads in the CDNA5 16-bit
// operand layout:
//   lanes 0-15  : row = l16, elems 0..7 -> K+0..7,  8..15 -> K+16..23
//   lanes 16-31 : row = l16, elems 0..7 -> K+8..15, 8..15 -> K+24..31
// Requires M % 128 == 0 and K % 32 == 0 (true for all call sites); N is
// guarded (x_dbl has N = 96).
// -----------------------------------------------------------------------------
__global__ void __launch_bounds__(256)
gemm_bf16_wmma(const float* __restrict__ A,
               const float* __restrict__ B,
               float* __restrict__ C,
               int M, int N, int K, int lda, int ldb, int ldc)
{
    __shared__ __align__(16) __bf16 sA[BM * LDSTR];   // 10.0 KB
    __shared__ __align__(16) __bf16 sB[BN * LDSTR];   //  5.0 KB

    const int tid  = threadIdx.x;
    const int lane = tid & 31;
    const int wave = tid >> 5;            // 0..7
    const int wm   = wave >> 1;           // 0..3 (M dir)
    const int wn   = wave & 1;            // 0..1 (N dir)
    const int half = lane >> 4;           // 0: lanes 0-15, 1: lanes 16-31
    const int l16  = lane & 15;

    const int blockM = blockIdx.y * BM;
    const int blockN = blockIdx.x * BN;

    // staging work split: A = 128 rows x 2 segs of 16 f32; B = 64 rows x 4 segs of 8
    const int arow = tid >> 1, aseg = tid & 1;
    const int brow = tid >> 2, bseg = tid & 3;
    const float* gA = A + (size_t)(blockM + arow) * lda + aseg * 16;
    const float* gB = B + (size_t)(blockN + brow) * ldb + bseg * 8;
    const bool bvalid = (blockN + brow) < N;

    v8f acc[2][2] = {};

    for (int k0 = 0; k0 < K; k0 += BK) {
        // ---- stage A tile (128 x 32) ----
        __builtin_prefetch(gA + k0 + BK, 0, 1);       // global_prefetch_b8
        v8bf a0 = cvt8(gA + k0);
        v8bf a1 = cvt8(gA + k0 + 8);
        *(v8bf*)&sA[arow * LDSTR + aseg * 16]     = a0;
        *(v8bf*)&sA[arow * LDSTR + aseg * 16 + 8] = a1;
        // ---- stage B tile (64 x 32), zero-fill rows beyond N ----
        v8bf b0 = {};
        if (bvalid) {
            __builtin_prefetch(gB + k0 + BK, 0, 1);
            b0 = cvt8(gB + k0);
        }
        *(v8bf*)&sB[brow * LDSTR + bseg * 8] = b0;
        __syncthreads();

        // ---- build fragments from LDS (16B ds loads) ----
        v16bf af[2], bfr[2];
#pragma unroll
        for (int mi = 0; mi < 2; ++mi) {
            int r = wm * 32 + mi * 16 + l16;
            v8bf lo = *(const v8bf*)&sA[r * LDSTR + half * 8];
            v8bf hi = *(const v8bf*)&sA[r * LDSTR + 16 + half * 8];
            af[mi] = __builtin_shufflevector(lo, hi,
                     0,1,2,3,4,5,6,7,8,9,10,11,12,13,14,15);
        }
#pragma unroll
        for (int ni = 0; ni < 2; ++ni) {
            int r = wn * 32 + ni * 16 + l16;
            v8bf lo = *(const v8bf*)&sB[r * LDSTR + half * 8];
            v8bf hi = *(const v8bf*)&sB[r * LDSTR + 16 + half * 8];
            bfr[ni] = __builtin_shufflevector(lo, hi,
                      0,1,2,3,4,5,6,7,8,9,10,11,12,13,14,15);
        }

        // ---- 2x2 WMMA ----
#pragma unroll
        for (int mi = 0; mi < 2; ++mi)
#pragma unroll
            for (int ni = 0; ni < 2; ++ni)
                acc[mi][ni] = __builtin_amdgcn_wmma_f32_16x16x32_bf16(
                    false, af[mi], false, bfr[ni], (short)0, acc[mi][ni],
                    false, false);
        __syncthreads();
    }

    // ---- store C: acc elem r -> row base + half*8 + r, col base + l16 ----
#pragma unroll
    for (int mi = 0; mi < 2; ++mi) {
#pragma unroll
        for (int ni = 0; ni < 2; ++ni) {
            int ccol = blockN + wn * 32 + ni * 16 + l16;
            if (ccol < N) {
                float* cp = C + (size_t)(blockM + wm * 32 + mi * 16 + half * 8) * ldc + ccol;
#pragma unroll
                for (int r = 0; r < 8; ++r) cp[(size_t)r * ldc] = acc[mi][ni][r];
            }
        }
    }
}

// -----------------------------------------------------------------------------
// Depthwise causal conv (width 4) over L, per channel, + bias + SiLU.
// xr is [NROWS, 2*DI]; channel data xc = xr[:, :DI]. Output u in (b,l,d).
// -----------------------------------------------------------------------------
__global__ void conv_silu(const float* __restrict__ xr,
                          const float* __restrict__ Wc,   // [DI, DCONV]
                          const float* __restrict__ bc,   // [DI]
                          float* __restrict__ u_t)        // [NROWS, DI]
{
    int idx = blockIdx.x * blockDim.x + threadIdx.x;
    if (idx >= NROWS * DI) return;
    int d   = idx % DI;
    int row = idx / DI;            // b*LSEQ + l
    int l   = row % LSEQ;
    int b   = row / LSEQ;

    float acc = bc[d];
#pragma unroll
    for (int j = 0; j < DCONV; ++j) {
        int ll = l - (DCONV - 1) + j;
        if (ll >= 0)
            acc += xr[(size_t)(b * LSEQ + ll) * (2 * DI) + d] * Wc[d * DCONV + j];
    }
    u_t[(size_t)row * DI + d] = acc * sigf(acc);
}

// delta = softplus(dtproj + b_dt), in place on [NROWS, DI]
__global__ void softplus_bias(float* __restrict__ v, const float* __restrict__ bdt)
{
    int idx = blockIdx.x * blockDim.x + threadIdx.x;
    if (idx >= NROWS * DI) return;
    float x = v[idx] + bdt[idx % DI];
    v[idx]  = (x > 20.0f) ? x : log1pf(__expf(x));
}

// -----------------------------------------------------------------------------
// Selective-scan recurrence. One thread per (b,d): h[16] in registers,
// sequential over L. B/C reads are wave-uniform (lanes of a wave share b).
// Fuses the u*D skip and the silu(res) gate so y2 feeds the final GEMM.
// -----------------------------------------------------------------------------
__global__ void ssm_scan(const float* __restrict__ delta,  // [NROWS, DI]
                         const float* __restrict__ u_t,    // [NROWS, DI]
                         const float* __restrict__ xdbl,   // [NROWS, 96] dt|B|C
                         const float* __restrict__ xr,     // [NROWS, 2*DI], res at +DI
                         const float* __restrict__ Alog,   // [DI, DSTATE]
                         const float* __restrict__ Dp,     // [DI]
                         float* __restrict__ y2)           // [NROWS, DI]
{
    int t = blockIdx.x * blockDim.x + threadIdx.x;
    if (t >= BATCH * DI) return;
    int b = t / DI, d = t % DI;

    float a[DSTATE], h[DSTATE];
#pragma unroll
    for (int n = 0; n < DSTATE; ++n) {
        a[n] = -__expf(Alog[d * DSTATE + n]);   // A = -exp(A_log)
        h[n] = 0.0f;
    }
    float wD = Dp[d];

    for (int l = 0; l < LSEQ; ++l) {
        size_t row = (size_t)(b * LSEQ + l);
        float dl  = delta[row * DI + d];
        float u   = u_t[row * DI + d];
        float res = xr[row * (2 * DI) + DI + d];
        const float* xd = xdbl + row * XPROJ_E;

        float du = dl * u;
        float y  = 0.0f;
#pragma unroll
        for (int n = 0; n < DSTATE; ++n) {
            float dA = __expf(dl * a[n]);               // exp(delta * A)
            h[n] = h[n] * dA + du * xd[DTRANK + n];     // h = dA*h + delta*B*u
            y += h[n] * xd[DTRANK + DSTATE + n];        // y = <h, C>
        }
        y += u * wD;                                    // + u * D
        y2[row * DI + d] = y * (res * sigf(res));       // * silu(res)
    }
}

// -----------------------------------------------------------------------------
extern "C" void kernel_launch(void* const* d_in, const int* in_sizes, int n_in,
                              void* d_out, int out_size, void* d_ws, size_t ws_size,
                              hipStream_t stream)
{
    const float* x      = (const float*)d_in[0];   // [B, L, DIM]
    const float* W_in   = (const float*)d_in[1];   // [2*DI, DIM]
    const float* W_conv = (const float*)d_in[2];   // [DI, 1, DCONV]
    const float* b_conv = (const float*)d_in[3];   // [DI]
    const float* W_xp   = (const float*)d_in[4];   // [96, DI]
    const float* W_dt   = (const float*)d_in[5];   // [DI, DTRANK]
    const float* b_dt   = (const float*)d_in[6];   // [DI]
    const float* A_log  = (const float*)d_in[7];   // [DI, DSTATE]
    const float* Dp     = (const float*)d_in[8];   // [DI]
    const float* W_out  = (const float*)d_in[9];   // [DIM, DI]
    float* out = (float*)d_out;                    // [B, L, DIM] f32

    // workspace partition (~85 MB)
    char* w = (char*)d_ws;
    float* xr    = (float*)w; w += (size_t)NROWS * 2 * DI   * sizeof(float); // 32 MB
    float* u_t   = (float*)w; w += (size_t)NROWS * DI       * sizeof(float); // 16 MB
    float* xdbl  = (float*)w; w += (size_t)NROWS * XPROJ_E  * sizeof(float); // .75 MB
    float* delta = (float*)w; w += (size_t)NROWS * DI       * sizeof(float); // 16 MB
    float* y2    = (float*)w; w += (size_t)NROWS * DI       * sizeof(float); // 16 MB

    const int BLK = 256;
    auto ggrid = [](int M, int N) { return dim3((unsigned)((N + BN - 1) / BN),
                                                (unsigned)(M / BM)); };

    // 1) xr = x @ W_in^T         (2048 x 4096 x 1024)
    gemm_bf16_wmma<<<ggrid(NROWS, 2 * DI), BLK, 0, stream>>>(
        x, W_in, xr, NROWS, 2 * DI, DIM, DIM, DIM, 2 * DI);

    // 2) u = silu(depthwise_conv(xc) + b_conv)   -> (b,l,d)
    conv_silu<<<(NROWS * DI + BLK - 1) / BLK, BLK, 0, stream>>>(xr, W_conv, b_conv, u_t);

    // 3) x_dbl = u @ W_xproj^T   (2048 x 96 x 2048)
    gemm_bf16_wmma<<<ggrid(NROWS, XPROJ_E), BLK, 0, stream>>>(
        u_t, W_xp, xdbl, NROWS, XPROJ_E, DI, DI, DI, XPROJ_E);

    // 4) dtproj = dt @ W_dt^T    (2048 x 2048 x 64), dt = x_dbl[:, :64] (lda=96)
    gemm_bf16_wmma<<<ggrid(NROWS, DI), BLK, 0, stream>>>(
        xdbl, W_dt, delta, NROWS, DI, DTRANK, XPROJ_E, DTRANK, DI);

    // 5) delta = softplus(dtproj + b_dt)
    softplus_bias<<<(NROWS * DI + BLK - 1) / BLK, BLK, 0, stream>>>(delta, b_dt);

    // 6) selective scan (fused u*D skip + silu(res) gate)
    ssm_scan<<<(BATCH * DI + BLK - 1) / BLK, BLK, 0, stream>>>(
        delta, u_t, xdbl, xr, A_log, Dp, y2);

    // 7) out = y2 @ W_out^T      (2048 x 1024 x 2048)
    gemm_bf16_wmma<<<ggrid(NROWS, DIM), BLK, 0, stream>>>(
        y2, W_out, out, NROWS, DIM, DI, DI, DI, DIM);
}
